// GraphSAGE_63788854280596
// MI455X (gfx1250) — matrix operand order repaired and checked by reading, not compiled, not run on hardware
//
#include <hip/hip_runtime.h>
#include <hip/hip_bf16.h>

#define GN_NODES 50000
#define GN_EDGES 800000

typedef __bf16 bf16;
typedef __attribute__((ext_vector_type(16))) __bf16 v16bf;
typedef __attribute__((ext_vector_type(8)))  __bf16 v8bf;
typedef __attribute__((ext_vector_type(4)))  __bf16 v4bf;
typedef __attribute__((ext_vector_type(8)))  float  v8f;

// ---------------------------------------------------------------- convert f32 -> bf16 (vectorized x4)
__global__ void gsage_cvt_bf16(const float* __restrict__ in, bf16* __restrict__ out, int n4) {
  int i = blockIdx.x * 256 + threadIdx.x;
  if (i < n4) {
    float4 f = ((const float4*)in)[i];
    v4bf b;
    b[0] = (bf16)f.x; b[1] = (bf16)f.y; b[2] = (bf16)f.z; b[3] = (bf16)f.w;
    ((v4bf*)out)[i] = b;
  }
}

// ---------------------------------------------------------------- CSR row_ptr from sorted rows
__global__ void gsage_rowptr(const int* __restrict__ rows, int* __restrict__ rp,
                             int n_nodes, int n_edges) {
  int i = blockIdx.x * 256 + threadIdx.x;
  if (i > n_nodes) return;
  int lo = 0, hi = n_edges;
  while (lo < hi) { int mid = (lo + hi) >> 1; if (rows[mid] < i) lo = mid + 1; else hi = mid; }
  rp[i] = lo;   // first edge with row >= i
}

// ---------------------------------------------------------------- SpMM: out[row,:] = sum_e val*feat[col,:]
// one block (256 thr) per destination node; edge chunk staged once in LDS, thread = channel
template <typename T>
__global__ __launch_bounds__(256)
void gsage_spmm(const T* __restrict__ feat, const int* __restrict__ cols,
                const float* __restrict__ vals, const int* __restrict__ rp,
                bf16* __restrict__ out) {
  __shared__ int   sc[256];
  __shared__ float sv[256];
  const int row = blockIdx.x;
  const int c   = threadIdx.x;
  const int e0 = rp[row], e1 = rp[row + 1];
  float acc = 0.f;
  for (int base = e0; base < e1; base += 256) {
    const int n = min(256, e1 - base);
    if (threadIdx.x < n) {
      sc[threadIdx.x] = cols[base + threadIdx.x];
      sv[threadIdx.x] = vals[base + threadIdx.x];
    }
    __syncthreads();
    for (int j = 0; j < n; ++j)
      acc += sv[j] * (float)feat[sc[j] * 256 + c];
    __syncthreads();
  }
  out[row * 256 + c] = (bf16)acc;
}

// ---------------------------------------------------------------- bf16 WMMA GEMM, software-pipelined
// C[M,NCOLS] = act( [A0|A1][M,KTOT] * B[KTOT,NCOLS] + bias ), A rows 256-wide bf16 (A1 for k>=256)
// block = 256 thr (8 waves), tile 128x128, BK=32; wave grid 2(M)x4(N), each wave 64x32.
#define LDS_A 40   // 32 + 8 pad (halves); 80B row stride -> 16B aligned b128 frags
#define LDS_B 40

template <int KTOT, int NCOLS, bool HAS_A1, bool OUT_F32, bool RELU>
__global__ __launch_bounds__(256)
void gsage_gemm_wmma(const bf16* __restrict__ A0, const bf16* __restrict__ A1,
                     const bf16* __restrict__ B, const float* __restrict__ bias,
                     void* __restrict__ Cout, int M) {
  constexpr int NT = KTOT / 32;
  __shared__ __align__(16) bf16 As[2][128 * LDS_A];
  __shared__ __align__(16) bf16 Bs[2][128 * LDS_B];

  const int tid   = threadIdx.x;
  const int lane  = tid & 31;
  const int wid   = tid >> 5;
  const int waveM = wid & 1;       // 0..1  -> 64 rows each
  const int waveN = wid >> 1;      // 0..3  -> 32 cols each
  const int m0 = blockIdx.x * 128;
  const int n0 = blockIdx.y * 128;
  const int half = lane >> 4;      // K-half select per ISA A/B layouts
  const int lm   = lane & 15;

  // staging coordinates (fixed per thread)
  const int ar   = tid >> 3, ach = tid & 7;          // A: row / 4-half chunk (x4 iters, +32 rows)
  const int bkr4 = (tid >> 5) * 4;                   // B: 4 K-rows starting here
  const int bnc  = (tid & 31) * 4;                   // B: 4 N-cols starting here

  // precomputed, branch-free A offsets (rows clamped; clamped rows never stored)
  unsigned aOff[4];
#pragma unroll
  for (int it = 0; it < 4; ++it) {
    int gm = m0 + ar + it * 32;
    gm = gm < M ? gm : M - 1;
    aOff[it] = (unsigned)(gm * 256 + ach * 4);
  }
  const bf16* bBase = B + bkr4 * NCOLS + n0 + bnc;

  v8f acc[4][2];
#pragma unroll
  for (int mt = 0; mt < 4; ++mt)
#pragma unroll
    for (int nt = 0; nt < 2; ++nt)
#pragma unroll
      for (int i = 0; i < 8; ++i) acc[mt][nt][i] = 0.f;

  // ---- register-load of tile kt (all k-dependent offsets are compile-time after unroll)
  uint2 areg[4];
  v4bf  brow[4];
  auto load_regs = [&](int kt) {
    const int k = kt * 32;
    const bf16* Ap = (HAS_A1 && k >= 256) ? A1 : A0;
    const int   kl = (HAS_A1 && k >= 256) ? k - 256 : k;
#pragma unroll
    for (int it = 0; it < 4; ++it)
      areg[it] = *(const uint2*)(Ap + aOff[it] + kl);
#pragma unroll
    for (int r = 0; r < 4; ++r)           // 4 consecutive K rows, 4 N cols each (8B coalesced)
      brow[r] = *(const v4bf*)(bBase + (k + r) * NCOLS);
  };
  // ---- LDS store of staged registers into buffer `buf` (B repacked transposed: Bs[n][k])
  auto store_lds = [&](int buf) {
#pragma unroll
    for (int it = 0; it < 4; ++it)
      *(uint2*)(&As[buf][(ar + it * 32) * LDS_A + ach * 4]) = areg[it];
#pragma unroll
    for (int j = 0; j < 4; ++j) {         // register-transpose 4x4, then 8B LDS stores
      v4bf colv;
      colv[0] = brow[0][j]; colv[1] = brow[1][j];
      colv[2] = brow[2][j]; colv[3] = brow[3][j];
      *(v4bf*)(&Bs[buf][(bnc + j) * LDS_B + bkr4]) = colv;
    }
  };
  // ---- WMMA compute on buffer `buf`
  auto compute = [&](int buf) {
    v16bf afrag[4], bfrag[2];
#pragma unroll
    for (int mt = 0; mt < 4; ++mt) {   // A: lane<16 -> K{0..7,16..23}; lane>=16 -> K{8..15,24..31}
      const bf16* p = &As[buf][(waveM * 64 + mt * 16 + lm) * LDS_A];
      v8bf lo = *(const v8bf*)(p + half * 8);
      v8bf hi = *(const v8bf*)(p + 16 + half * 8);
      afrag[mt] = __builtin_shufflevector(lo, hi, 0,1,2,3,4,5,6,7,8,9,10,11,12,13,14,15);
    }
#pragma unroll
    for (int nt = 0; nt < 2; ++nt) {   // B: lane n = lm; lane<16 -> K{0..7,16..23}; lane>=16 -> K{8..15,24..31}
      const bf16* p = &Bs[buf][(waveN * 32 + nt * 16 + lm) * LDS_B];
      v8bf lo = *(const v8bf*)(p + half * 8);
      v8bf hi = *(const v8bf*)(p + 16 + half * 8);
      bfrag[nt] = __builtin_shufflevector(lo, hi, 0,1,2,3,4,5,6,7,8,9,10,11,12,13,14,15);
    }
#pragma unroll
    for (int mt = 0; mt < 4; ++mt)
#pragma unroll
      for (int nt = 0; nt < 2; ++nt)
        acc[mt][nt] = __builtin_amdgcn_wmma_f32_16x16x32_bf16(
            false, afrag[mt], false, bfrag[nt], (short)0, acc[mt][nt], false, false);
  };

  // ---- pipeline: prologue stages tile 0, then overlap global loads with WMMA
  load_regs(0);
  store_lds(0);
  __syncthreads();
#pragma unroll
  for (int kt = 0; kt < NT; ++kt) {
    if (kt + 1 < NT) load_regs(kt + 1);
    compute(kt & 1);
    if (kt + 1 < NT) store_lds((kt + 1) & 1);
    __syncthreads();
  }

  // ---- epilogue: C layout -> VGPR i: M = half*8 + i ; N = lm
#pragma unroll
  for (int mt = 0; mt < 4; ++mt) {
#pragma unroll
    for (int nt = 0; nt < 2; ++nt) {
      int col = n0 + waveN * 32 + nt * 16 + lm;
      float bv = bias[col];
#pragma unroll
      for (int i = 0; i < 8; ++i) {
        int row = m0 + waveM * 64 + mt * 16 + half * 8 + i;
        if (row < M) {
          float v = acc[mt][nt][i] + bv;
          if (RELU) v = fmaxf(v, 0.f);
          if (OUT_F32) ((float*)Cout)[(size_t)row * NCOLS + col] = v;
          else         ((bf16*)Cout)[(size_t)row * NCOLS + col] = (bf16)v;
        }
      }
    }
  }
}

// ---------------------------------------------------------------- row-wise L2 normalize (wave per row)
__global__ void gsage_l2norm(const float* __restrict__ in, float* __restrict__ out, int n_rows) {
  int gw   = (blockIdx.x * blockDim.x + threadIdx.x) >> 5;
  int lane = threadIdx.x & 31;
  if (gw >= n_rows) return;
  const float* r = in + (size_t)gw * 128;
  float v0 = r[lane], v1 = r[lane + 32], v2 = r[lane + 64], v3 = r[lane + 96];
  float s = v0 * v0 + v1 * v1 + v2 * v2 + v3 * v3;
#pragma unroll
  for (int off = 16; off > 0; off >>= 1) s += __shfl_xor(s, off, 32);
  float inv = 1.0f / fmaxf(sqrtf(s), 1e-12f);
  float* o = out + (size_t)gw * 128;
  o[lane] = v0 * inv; o[lane + 32] = v1 * inv; o[lane + 64] = v2 * inv; o[lane + 96] = v3 * inv;
}

// ---------------------------------------------------------------- driver
extern "C" void kernel_launch(void* const* d_in, const int* in_sizes, int n_in,
                              void* d_out, int out_size, void* d_ws, size_t ws_size,
                              hipStream_t stream) {
  (void)in_sizes; (void)n_in; (void)out_size; (void)ws_size;
  const float* x        = (const float*)d_in[0];
  const int*   adj_rows = (const int*)  d_in[1];
  const int*   adj_cols = (const int*)  d_in[2];
  const float* adj_vals = (const float*)d_in[3];
  const float* Wa0 = (const float*)d_in[4];
  const float* ba0 = (const float*)d_in[5];
  const float* Wa1 = (const float*)d_in[6];
  const float* ba1 = (const float*)d_in[7];
  const float* Wl0 = (const float*)d_in[8];
  const float* bl0 = (const float*)d_in[9];
  const float* Wl1 = (const float*)d_in[10];
  const float* bl1 = (const float*)d_in[11];

  char* ws = (char*)d_ws;
  size_t off = 0;
  auto alloc = [&](size_t bytes) -> void* {
    void* p = ws + off;
    off = (off + bytes + 255) & ~(size_t)255;
    return p;
  };
  bf16* xb     = (bf16*)alloc((size_t)GN_NODES * 256 * 2);   // x in bf16
  bf16* aggb   = (bf16*)alloc((size_t)GN_NODES * 256 * 2);   // spmm output (reused)
  bf16* aggA   = (bf16*)alloc((size_t)GN_NODES * 256 * 2);   // relu(gemm) output (reused)
  bf16* h      = (bf16*)alloc((size_t)GN_NODES * 256 * 2);   // hidden features
  float* outr  = (float*)alloc((size_t)GN_NODES * 128 * 4);  // pre-normalize output
  int*  rowptr = (int*)  alloc((size_t)(GN_NODES + 1) * 4);
  bf16* wa0b   = (bf16*)alloc((size_t)256 * 256 * 2);
  bf16* wa1b   = (bf16*)alloc((size_t)256 * 256 * 2);
  bf16* wl0b   = (bf16*)alloc((size_t)512 * 256 * 2);
  bf16* wl1b   = (bf16*)alloc((size_t)512 * 128 * 2);

  // 1) precision conversions (x4 vectorized)
  gsage_cvt_bf16<<<(GN_NODES * 64 + 255) / 256, 256, 0, stream>>>(x,   xb,   GN_NODES * 64);
  gsage_cvt_bf16<<<(256 * 64 + 255) / 256,      256, 0, stream>>>(Wa0, wa0b, 256 * 64);
  gsage_cvt_bf16<<<(256 * 64 + 255) / 256,      256, 0, stream>>>(Wa1, wa1b, 256 * 64);
  gsage_cvt_bf16<<<(512 * 64 + 255) / 256,      256, 0, stream>>>(Wl0, wl0b, 512 * 64);
  gsage_cvt_bf16<<<(512 * 32 + 255) / 256,      256, 0, stream>>>(Wl1, wl1b, 512 * 32);

  // 2) CSR row pointers from sorted adj_rows
  gsage_rowptr<<<(GN_NODES + 1 + 255) / 256, 256, 0, stream>>>(adj_rows, rowptr, GN_NODES, GN_EDGES);

  // 3) spmm(x) -> aggb (gather f32 x, L2-resident)
  gsage_spmm<float><<<GN_NODES, 256, 0, stream>>>(x, adj_cols, adj_vals, rowptr, aggb);

  const int MB = (GN_NODES + 127) / 128;  // 391
  // 4) aggA = relu(aggb @ Wa0 + ba0)
  gsage_gemm_wmma<256, 256, false, false, true><<<dim3(MB, 2), 256, 0, stream>>>(
      aggb, nullptr, wa0b, ba0, aggA, GN_NODES);
  // 5) h = relu([x | aggA] @ Wl0 + bl0)
  gsage_gemm_wmma<512, 256, true, false, true><<<dim3(MB, 2), 256, 0, stream>>>(
      xb, aggA, wl0b, bl0, h, GN_NODES);
  // 6) spmm(h) -> aggb (gather bf16: halves the gather traffic)
  gsage_spmm<bf16><<<GN_NODES, 256, 0, stream>>>(h, adj_cols, adj_vals, rowptr, aggb);
  // 7) aggA = relu(aggb @ Wa1 + ba1)
  gsage_gemm_wmma<256, 256, false, false, true><<<dim3(MB, 2), 256, 0, stream>>>(
      aggb, nullptr, wa1b, ba1, aggA, GN_NODES);
  // 8) outr = [h | aggA] @ Wl1 + bl1  (f32 out)
  gsage_gemm_wmma<512, 128, true, true, false><<<dim3(MB, 1), 256, 0, stream>>>(
      h, aggA, wl1b, bl1, outr, GN_NODES);
  // 9) row-wise L2 normalize -> d_out
  gsage_l2norm<<<(GN_NODES * 32 + 255) / 256, 256, 0, stream>>>(outr, (float*)d_out, GN_NODES);
}
